// LocationSensitiveAttention_9552007266553
// MI455X (gfx1250) — matrix operand order repaired and checked by reading, not compiled, not run on hardware
//
#include <hip/hip_runtime.h>
#include <hip/hip_bf16.h>

#define HID  512
#define LSEQ 4096
#define BATCH 32

typedef __attribute__((ext_vector_type(16))) __bf16 v16bf;
typedef __attribute__((ext_vector_type(8)))  float  v8f;
typedef __attribute__((ext_vector_type(4)))  float  f4;

// ---------------- prep: split Wv into bf16 hi/lo tables ----------------
__global__ void wvsplit_kernel(const float* __restrict__ Wv,
                               __bf16* __restrict__ hi, __bf16* __restrict__ lo) {
  int i = blockIdx.x * blockDim.x + threadIdx.x;      // 0..262143
  float w = Wv[i];
  __bf16 h = (__bf16)w;
  hi[i] = h;
  lo[i] = (__bf16)(w - (float)h);
}

// ---------------- q projection: q[b,o] = sum_h query[b,h]*Wq[o,h] ----------------
__global__ void qproj_kernel(const float* __restrict__ query,
                             const float* __restrict__ Wq,
                             float* __restrict__ qws) {
  int i = blockIdx.x * blockDim.x + threadIdx.x;      // 0..16383
  int b = i >> 9, o = i & (HID - 1);
  const float* qr = query + b * HID;
  const float* wr = Wq + (size_t)o * HID;
  float acc = 0.f;
  for (int h = 0; h < HID; ++h) acc += qr[h] * wr[h];
  qws[i] = acc;
}

// ---------------- main fused kernel: v-GEMM (bf16x3 WMMA) + conv + tanh + score ----------------
__global__ __launch_bounds__(128) void score_kernel(
    const float*  __restrict__ value,   // (B,L,H)
    const float*  __restrict__ la,      // (B,L)
    const __bf16* __restrict__ wvhi,    // (H,H) bf16 hi
    const __bf16* __restrict__ wvlo,    // (H,H) bf16 lo
    const float*  __restrict__ qws,     // (B,H)
    const float*  __restrict__ Wc,      // (H,3)
    const float*  __restrict__ bc,      // (H)
    const float*  __restrict__ Ws,      // (H)
    const float*  __restrict__ bs,      // (1)
    const float*  __restrict__ bias,    // (H)
    float*        __restrict__ s_out)   // (B,L) sigmoid(score)
{
  const int blk    = blockIdx.x;         // one per 16 rows of (B*L)
  const int rbase  = blk * 16;
  const int b      = rbase >> 12;        // / LSEQ
  const int lbase  = rbase & (LSEQ - 1);
  const int tid    = threadIdx.x;
  const int wave   = tid >> 5;           // 0..3
  const int lane   = tid & 31;
  const int laneLo = lane & 15;
  const int laneHi = lane >> 4;          // 0 or 1
  const int colBase = wave * 128;        // 8 tiles of 16 cols per wave

  const v8f vz = {0.f,0.f,0.f,0.f,0.f,0.f,0.f,0.f};
  v8f C[8];
#pragma unroll
  for (int t = 0; t < 8; ++t) C[t] = vz;

  const float* Arow = value + (size_t)(rbase + laneLo) * HID;
  const int kOffA = laneHi * 16;         // A frag: lanes 0-15 K 0-15, lanes 16-31 K 16-31

  for (int k0 = 0; k0 < HID; k0 += 32) {
    // ---- load 16 f32 of this wave's A fragment, split into bf16 hi/lo ----
    union { f4 v[4]; float f[16]; } af;
    af.v[0] = *(const f4*)(Arow + k0 + kOffA);
    af.v[1] = *(const f4*)(Arow + k0 + kOffA + 4);
    af.v[2] = *(const f4*)(Arow + k0 + kOffA + 8);
    af.v[3] = *(const f4*)(Arow + k0 + kOffA + 12);
    v16bf ahi, alo;
#pragma unroll
    for (int i = 0; i < 16; ++i) {
      float x = af.f[i];
      __bf16 h = (__bf16)x;
      ahi[i] = h;
      alo[i] = (__bf16)(x - (float)h);
    }
#pragma unroll
    for (int t = 0; t < 8; ++t) {
      const size_t o = (size_t)(colBase + t * 16 + laneLo);
      const size_t boff = o * HID + k0 + laneHi * 16;   // 16 consecutive K per lane
      v16bf bhi = *(const v16bf*)(wvhi + boff);
      v16bf blo = *(const v16bf*)(wvlo + boff);
      // C += Alo*Bhi + Ahi*Blo + Ahi*Bhi   (f32 accumulate; lo*lo omitted, ~2^-18 rel)
      C[t] = __builtin_amdgcn_wmma_f32_16x16x32_bf16(false, alo, false, bhi,
                                                     (short)0, C[t], false, false);
      C[t] = __builtin_amdgcn_wmma_f32_16x16x32_bf16(false, ahi, false, blo,
                                                     (short)0, C[t], false, false);
      C[t] = __builtin_amdgcn_wmma_f32_16x16x32_bf16(false, ahi, false, bhi,
                                                     (short)0, C[t], false, false);
    }
  }

  // ---- epilogue: conv(last_attn) + bias + q, tanh, dot with Ws ----
  const float* laB = la + (size_t)b * LSEQ;
  float la0[8], lam[8], lap[8];
#pragma unroll
  for (int r = 0; r < 8; ++r) {
    int m = r + laneHi * 8;              // local row for C VGPR r
    int l = lbase + m;
    la0[r] = laB[l];
    lam[r] = (l > 0)        ? laB[l - 1] : 0.f;
    lap[r] = (l < LSEQ - 1) ? laB[l + 1] : 0.f;
  }

  float sAcc[8];
#pragma unroll
  for (int r = 0; r < 8; ++r) sAcc[r] = 0.f;

#pragma unroll
  for (int t = 0; t < 8; ++t) {
    const int o = colBase + t * 16 + laneLo;
    const float wsw = Ws[o];
    const float cb  = bc[o] + bias[o] + qws[b * HID + o];
    const float wc0 = Wc[o * 3 + 0], wc1 = Wc[o * 3 + 1], wc2 = Wc[o * 3 + 2];
#pragma unroll
    for (int r = 0; r < 8; ++r) {
      float e = C[t][r] + cb + lam[r] * wc0 + la0[r] * wc1 + lap[r] * wc2;
      sAcc[r] += wsw * tanhf(e);
    }
  }

  // reduce across the 16 lanes of each half-wave (xor over bits 0..3 keeps halves separate)
#pragma unroll
  for (int off = 1; off < 16; off <<= 1)
#pragma unroll
    for (int r = 0; r < 8; ++r)
      sAcc[r] += __shfl_xor(sAcc[r], off, 32);

  __shared__ float sred[4][16];
  if (laneLo == 0) {                      // lanes 0 (rows 0-7) and 16 (rows 8-15)
#pragma unroll
    for (int r = 0; r < 8; ++r) sred[wave][laneHi * 8 + r] = sAcc[r];
  }
  __syncthreads();
  if (tid < 16) {
    float tot = sred[0][tid] + sred[1][tid] + sred[2][tid] + sred[3][tid] + bs[0];
    float sg  = 1.f / (1.f + __expf(-tot));
    s_out[(size_t)b * LSEQ + lbase + tid] = sg;
  }
}

// ---------------- deterministic per-batch sum of sigmoids ----------------
__global__ void ssum_kernel(const float* __restrict__ s, float* __restrict__ sums) {
  __shared__ float red[256];
  const int b = blockIdx.x, tid = threadIdx.x;
  float a = 0.f;
  for (int l = tid; l < LSEQ; l += 256) a += s[(size_t)b * LSEQ + l];
  red[tid] = a;
  __syncthreads();
  for (int off = 128; off > 0; off >>= 1) {
    if (tid < off) red[tid] += red[tid + off];
    __syncthreads();
  }
  if (tid == 0) sums[b] = red[0];
}

// ---------------- attn output: attn = s / sum ----------------
__global__ void attn_kernel(const float* __restrict__ s, const float* __restrict__ sums,
                            float* __restrict__ out_attn) {
  int i = blockIdx.x * blockDim.x + threadIdx.x;      // 0..131071
  int b = i >> 12;
  out_attn[i] = s[i] / sums[b];
}

// ---------------- context stage 1: partial[b,lc,h] = sum_{l in chunk} s*value ----------------
__global__ void ctx1_kernel(const float* __restrict__ value, const float* __restrict__ s,
                            float* __restrict__ part) {
  const int b  = blockIdx.x;
  const int hc = blockIdx.y;               // 0..1
  const int lc = blockIdx.z;               // 0..15
  const int h  = hc * 256 + threadIdx.x;
  const int l0 = lc * (LSEQ / 16);
  float acc = 0.f;
  const float* vb = value + ((size_t)b * LSEQ + l0) * HID + h;
  const float* sb = s + (size_t)b * LSEQ + l0;
  for (int l = 0; l < LSEQ / 16; ++l) acc += sb[l] * vb[(size_t)l * HID];
  part[((size_t)b * 16 + lc) * HID + h] = acc;
}

// ---------------- context stage 2: context[b,h] = (1/sum) * sum_lc partial ----------------
__global__ void ctx2_kernel(const float* __restrict__ part, const float* __restrict__ sums,
                            float* __restrict__ out_ctx) {
  const int b = blockIdx.x;
  const int h = blockIdx.y * 256 + threadIdx.x;
  float acc = 0.f;
#pragma unroll
  for (int lc = 0; lc < 16; ++lc) acc += part[((size_t)b * 16 + lc) * HID + h];
  out_ctx[(size_t)b * HID + h] = acc / sums[b];
}

extern "C" void kernel_launch(void* const* d_in, const int* in_sizes, int n_in,
                              void* d_out, int out_size, void* d_ws, size_t ws_size,
                              hipStream_t stream) {
  const float* query = (const float*)d_in[0];   // (32,1,512)
  const float* value = (const float*)d_in[1];   // (32,4096,512)
  const float* la    = (const float*)d_in[2];   // (32,4096)
  const float* Wc    = (const float*)d_in[3];   // (512,1,3)
  const float* bc    = (const float*)d_in[4];   // (512)
  const float* Wq    = (const float*)d_in[5];   // (512,512)
  const float* Wv    = (const float*)d_in[6];   // (512,512)
  const float* Ws    = (const float*)d_in[7];   // (1,512)
  const float* bs    = (const float*)d_in[8];   // (1)
  const float* bias  = (const float*)d_in[9];   // (512)

  char* ws = (char*)d_ws;
  // workspace layout (bytes)
  float*  qws   = (float*)(ws + 0);                         //  65536 B : (32,512)
  float*  s_buf = (float*)(ws + 65536);                     // 524288 B : (32,4096)
  float*  sums  = (float*)(ws + 65536 + 524288);            //    128 B : (32)
  __bf16* wvhi  = (__bf16*)(ws + 590080);                   // 524288 B : (512,512) bf16
  __bf16* wvlo  = (__bf16*)(ws + 590080 + 524288);          // 524288 B
  float*  part  = (float*)(ws + 590080);                    // reuses wv region after score: (32,16,512) f32

  float* out_ctx  = (float*)d_out;                          // (32,512)
  float* out_attn = (float*)d_out + BATCH * HID;            // (32,4096)

  wvsplit_kernel<<<(HID * HID) / 256, 256, 0, stream>>>(Wv, wvhi, wvlo);
  qproj_kernel<<<(BATCH * HID) / 256, 256, 0, stream>>>(query, Wq, qws);

  score_kernel<<<(BATCH * LSEQ) / 16, 128, 0, stream>>>(
      value, la, wvhi, wvlo, qws, Wc, bc, Ws, bs, bias, s_buf);

  ssum_kernel<<<BATCH, 256, 0, stream>>>(s_buf, sums);
  attn_kernel<<<(BATCH * LSEQ) / 256, 256, 0, stream>>>(s_buf, sums, out_attn);

  ctx1_kernel<<<dim3(BATCH, 2, 16), 256, 0, stream>>>(value, s_buf, part);
  ctx2_kernel<<<dim3(BATCH, 2), 256, 0, stream>>>(part, sums, out_ctx);
}
